// Part_Dependency_9887014716016
// MI455X (gfx1250) — matrix operand order repaired and checked by reading, not compiled, not run on hardware
//
#include <hip/hip_runtime.h>
#include <hip/hip_bf16.h>

// ---------------------------------------------------------------------------
// Problem constants (from reference setup_inputs)
// ---------------------------------------------------------------------------
#define BATCH 4
#define HCH   128          // h
#define C2    256          // 2*h
#define HH    96
#define WW    96
#define HWSZ  (HH * WW)    // 9216
#define KRED  (C2 * 9)     // 2304  (GEMM reduction dim, reordered as k*C+c)

// GEMM tiling
#define BM 128
#define BN 128
#define BK 32
#define LDT 40             // padded LDS row stride in bf16 elements (80B, 16B-aligned)

typedef __attribute__((ext_vector_type(16))) __bf16 bf16x16;
typedef __attribute__((ext_vector_type(8)))  float  floatx8;

union Frag32 { float4 f4[2]; bf16x16 v; };

// CDNA5 async global->LDS path (ASYNCcnt), if the toolchain declares it.
#if defined(__has_builtin)
#if __has_builtin(__builtin_amdgcn_global_load_async_to_lds_b128) && \
    __has_builtin(__builtin_amdgcn_s_wait_asynccnt)
#define USE_ASYNC_LDS 1
#endif
#endif

#ifdef USE_ASYNC_LDS
typedef int v4i __attribute__((vector_size(16)));
typedef __attribute__((address_space(1))) v4i* g4ptr;   // global (HIP "__device__")
typedef __attribute__((address_space(3))) v4i* l4ptr;   // LDS    (HIP "__shared__")
__device__ __forceinline__ g4ptr to_g4(const void* p) {
    return (g4ptr)(unsigned long long)(size_t)p;
}
__device__ __forceinline__ l4ptr to_l4(const void* p) {
    // generic LDS pointer: low 32 bits are the LDS byte offset (ISA aperture rule)
    return (l4ptr)(unsigned int)(size_t)p;
}
#endif

__device__ __forceinline__ unsigned short f2bf(float f) {
    unsigned int u = __float_as_uint(f);
    unsigned int r = (u + 0x7FFFu + ((u >> 16) & 1u)) >> 16;   // RNE
    return (unsigned short)r;
}

__device__ __forceinline__ float sigmoidf_(float x) {
    return 1.0f / (1.0f + __expf(-x));
}

// ---------------------------------------------------------------------------
// concat: x[b][0:128]=pA, x[b][128:256]=pB   (NCHW f32)
// ---------------------------------------------------------------------------
__global__ void concat_kernel(const float* __restrict__ pA,
                              const float* __restrict__ pB,
                              float* __restrict__ x) {
    size_t idx = (size_t)blockIdx.x * blockDim.x + threadIdx.x;   // over B*HCH*HWSZ
    size_t per = (size_t)HCH * HWSZ;
    size_t b   = idx / per;
    size_t rem = idx % per;
    x[(b * C2) * HWSZ + rem]       = pA[idx];
    x[(b * C2 + HCH) * HWSZ + rem] = pB[idx];
}

// ---------------------------------------------------------------------------
// Weight conversion f32 -> bf16 with reduction-dim permutation:
//   dst[o][k*C + c] = bf16(src[o][c*9 + k])
// ---------------------------------------------------------------------------
__global__ void cvt_w_kernel(const float* __restrict__ src,
                             unsigned short* __restrict__ dst, int O, int C) {
    int i = blockIdx.x * blockDim.x + threadIdx.x;     // over O*C*9
    if (i >= O * C * 9) return;
    int o = i / (C * 9);
    int r = i % (C * 9);
    int c = r / 9;
    int k = r % 9;
    dst[(size_t)o * C * 9 + k * C + c] = f2bf(src[i]);
}

// ---------------------------------------------------------------------------
// Offset conv: 3x3, pad 1, C in channels -> 27 out channels, f32 direct
// grid (HWSZ/256, 27, B), block 256
// ---------------------------------------------------------------------------
__global__ __launch_bounds__(256)
void offconv_kernel(const float* __restrict__ x, const float* __restrict__ wt,
                    const float* __restrict__ bias, float* __restrict__ om, int C) {
    int hw = blockIdx.x * blockDim.x + threadIdx.x;
    int oc = blockIdx.y;
    int b  = blockIdx.z;
    int h = hw / WW, w = hw % WW;
    float acc = bias[oc];
    const float* xb = x + (size_t)b * C * HWSZ;
    const float* wo = wt + (size_t)oc * C * 9;
    for (int c = 0; c < C; ++c) {
        const float* xc = xb + (size_t)c * HWSZ;
        const float* wc = wo + c * 9;
        #pragma unroll
        for (int ky = 0; ky < 3; ++ky) {
            int y = h + ky - 1;
            if (y < 0 || y >= HH) continue;
            #pragma unroll
            for (int kx = 0; kx < 3; ++kx) {
                int xw = w + kx - 1;
                if (xw < 0 || xw >= WW) continue;
                acc = fmaf(xc[y * WW + xw], wc[ky * 3 + kx], acc);
            }
        }
    }
    om[((size_t)b * 27 + oc) * HWSZ + hw] = acc;
}

// ---------------------------------------------------------------------------
// DCNv2 bilinear sampling + mask -> im2col matrix S[n][k*C + c] bf16
// Contiguous 256-channel run per (n,k): packed 2x-bf16 32-bit stores.
// One batch per launch. grid (HWSZ/256, 9), block 256
// ---------------------------------------------------------------------------
__global__ __launch_bounds__(256)
void sample_kernel(const float* __restrict__ x, const float* __restrict__ om,
                   unsigned short* __restrict__ S, int C, int b) {
    int hw = blockIdx.x * blockDim.x + threadIdx.x;
    int k  = blockIdx.y;                  // 0..8
    int h = hw / WW, w = hw % WW;

    const float* omb = om + (size_t)b * 27 * HWSZ;
    float dy   = omb[(2 * k)     * HWSZ + hw];
    float dx   = omb[(2 * k + 1) * HWSZ + hw];
    float mval = sigmoidf_(omb[(18 + k) * HWSZ + hw]);

    float yy = (float)(h - 1 + (k / 3)) + dy;
    float xx = (float)(w - 1 + (k % 3)) + dx;
    float y0f = floorf(yy), x0f = floorf(xx);
    float wy1 = yy - y0f,  wx1 = xx - x0f;
    int y0 = (int)y0f, x0 = (int)x0f;
    int y1 = y0 + 1,   x1 = x0 + 1;

    bool vy0 = (y0 >= 0) && (y0 < HH);
    bool vy1 = (y1 >= 0) && (y1 < HH);
    bool vx0 = (x0 >= 0) && (x0 < WW);
    bool vx1 = (x1 >= 0) && (x1 < WW);
    int y0c = min(max(y0, 0), HH - 1), y1c = min(max(y1, 0), HH - 1);
    int x0c = min(max(x0, 0), WW - 1), x1c = min(max(x1, 0), WW - 1);

    float w00 = (vy0 && vx0) ? (1.f - wy1) * (1.f - wx1) : 0.f;
    float w01 = (vy0 && vx1) ? (1.f - wy1) * wx1         : 0.f;
    float w10 = (vy1 && vx0) ? wy1 * (1.f - wx1)         : 0.f;
    float w11 = (vy1 && vx1) ? wy1 * wx1                 : 0.f;
    w00 *= mval; w01 *= mval; w10 *= mval; w11 *= mval;

    int i00 = y0c * WW + x0c, i01 = y0c * WW + x1c;
    int i10 = y1c * WW + x0c, i11 = y1c * WW + x1c;

    const float* xb = x + (size_t)b * C * HWSZ;
    unsigned int* Srow = (unsigned int*)(S + (size_t)hw * (C * 9) + (size_t)k * C);
    for (int c = 0; c < C; c += 2) {
        const float* xc0 = xb + (size_t)c * HWSZ;
        const float* xc1 = xc0 + HWSZ;
        float v0 = xc0[i00] * w00 + xc0[i01] * w01 + xc0[i10] * w10 + xc0[i11] * w11;
        float v1 = xc1[i00] * w00 + xc1[i01] * w01 + xc1[i10] * w10 + xc1[i11] * w11;
        Srow[c >> 1] = (unsigned int)f2bf(v0) | ((unsigned int)f2bf(v1) << 16);
    }
}

// ---------------------------------------------------------------------------
// WMMA bf16 GEMM:  Y[o][n] = sum_r W[o][r] * S[n][r]
// W: [O][K] bf16 row-major;  S: [N][K] bf16 row-major (K contiguous per n)
// block 256 thr (8 waves, 2x4), tile 128x128, k-step 32, double-buffered LDS,
// staged via CDNA5 async global->LDS DMA (ASYNCcnt) when available.
// grid (N/BN, O/BM)
// ---------------------------------------------------------------------------
__global__ __launch_bounds__(256)
void gemm_wmma_kernel(const unsigned short* __restrict__ Wm,
                      const unsigned short* __restrict__ S,
                      float* __restrict__ Y, int O, int K, int N) {
    __shared__ __align__(16) unsigned short As[2][BM * LDT];
    __shared__ __align__(16) unsigned short Bs[2][BN * LDT];

    const int m0   = blockIdx.y * BM;
    const int n0   = blockIdx.x * BN;
    const int tid  = threadIdx.x;
    const int wave = tid >> 5;
    const int lane = tid & 31;
    const int wm   = wave >> 2;          // 0..1
    const int wn   = wave & 3;           // 0..3
    const int mlane = lane & 15;
    const int mhalf = lane >> 4;

    // staging indices: thread t handles float4 slots {t, t+256}; r=f>>2, c=f&3
    const int r0 = tid >> 2,          c0 = tid & 3;
    const int r1 = (tid + 256) >> 2,  c1 = tid & 3;

    floatx8 acc[4][2];
    #pragma unroll
    for (int i = 0; i < 4; ++i)
        #pragma unroll
        for (int j = 0; j < 2; ++j)
            acc[i][j] = (floatx8)0.0f;

#ifdef USE_ASYNC_LDS
    // ---- CDNA5 async LDS-DMA staging (ASYNCcnt) -------------------------
    auto stage_async = [&](int k0, int buf) {
        __builtin_amdgcn_global_load_async_to_lds_b128(
            to_g4(Wm + (size_t)(m0 + r0) * K + k0 + c0 * 8),
            to_l4(&As[buf][r0 * LDT + c0 * 8]), 0, 0);
        __builtin_amdgcn_global_load_async_to_lds_b128(
            to_g4(Wm + (size_t)(m0 + r1) * K + k0 + c1 * 8),
            to_l4(&As[buf][r1 * LDT + c1 * 8]), 0, 0);
        __builtin_amdgcn_global_load_async_to_lds_b128(
            to_g4(S + (size_t)(n0 + r0) * K + k0 + c0 * 8),
            to_l4(&Bs[buf][r0 * LDT + c0 * 8]), 0, 0);
        __builtin_amdgcn_global_load_async_to_lds_b128(
            to_g4(S + (size_t)(n0 + r1) * K + k0 + c1 * 8),
            to_l4(&Bs[buf][r1 * LDT + c1 * 8]), 0, 0);
    };
    stage_async(0, 0);
#else
    // ---- register-prefetch staging fallback -----------------------------
    float4 ra0, ra1, rb0, rb1;
    auto gload = [&](int k0) {
        ra0 = *(const float4*)(Wm + (size_t)(m0 + r0) * K + k0 + c0 * 8);
        ra1 = *(const float4*)(Wm + (size_t)(m0 + r1) * K + k0 + c1 * 8);
        rb0 = *(const float4*)(S + (size_t)(n0 + r0) * K + k0 + c0 * 8);
        rb1 = *(const float4*)(S + (size_t)(n0 + r1) * K + k0 + c1 * 8);
    };
    auto lstore = [&](int buf) {
        *(float4*)(&As[buf][r0 * LDT + c0 * 8]) = ra0;
        *(float4*)(&As[buf][r1 * LDT + c1 * 8]) = ra1;
        *(float4*)(&Bs[buf][r0 * LDT + c0 * 8]) = rb0;
        *(float4*)(&Bs[buf][r1 * LDT + c1 * 8]) = rb1;
    };
    gload(0);
#endif

    int buf = 0;
    for (int k0 = 0; k0 < K; k0 += BK) {
        const bool has_next = (k0 + BK) < K;

#ifdef USE_ASYNC_LDS
        if (has_next) {
            stage_async(k0 + BK, buf ^ 1);
            // oldest 4 ops (current buf) complete; next buf's 4 may stay in flight
            __builtin_amdgcn_s_wait_asynccnt(4);
        } else {
            __builtin_amdgcn_s_wait_asynccnt(0);
        }
        __syncthreads();
#else
        lstore(buf);
        __syncthreads();
        if (has_next) gload(k0 + BK);   // in flight during WMMA below
#endif

        // B fragments: lane l -> n = 16*jt + (l&15), K = {8*half..+7, 16+8*half..+7}
        Frag32 bfr[2];
        #pragma unroll
        for (int j = 0; j < 2; ++j) {
            const unsigned short* p = &Bs[buf][(wn * 32 + j * 16 + mlane) * LDT + 8 * mhalf];
            bfr[j].f4[0] = *(const float4*)(p);
            bfr[j].f4[1] = *(const float4*)(p + 16);
        }
        // A fragments: lane l -> m = l&15, K = {8*half..+7, 16+8*half..+7}
        #pragma unroll
        for (int i = 0; i < 4; ++i) {
            Frag32 afr;
            const unsigned short* p = &As[buf][(wm * 64 + i * 16 + mlane) * LDT];
            afr.f4[0] = *(const float4*)(p + 8 * mhalf);
            afr.f4[1] = *(const float4*)(p + 16 + 8 * mhalf);
            #pragma unroll
            for (int j = 0; j < 2; ++j) {
                acc[i][j] = __builtin_amdgcn_wmma_f32_16x16x32_bf16(
                    false, afr.v, false, bfr[j].v, (short)0, acc[i][j], false, false);
            }
        }

#ifdef USE_ASYNC_LDS
        __syncthreads();   // all ds reads of buf done before async engine rewrites it
#endif
        buf ^= 1;
    }

    // C/D layout: lane l, VGPR v -> M = v + 8*(l>>4), N = l&15
    #pragma unroll
    for (int i = 0; i < 4; ++i) {
        #pragma unroll
        for (int j = 0; j < 2; ++j) {
            int mBase = m0 + wm * 64 + i * 16 + 8 * mhalf;
            int nIdx  = n0 + wn * 32 + j * 16 + mlane;
            #pragma unroll
            for (int v = 0; v < 8; ++v)
                Y[(size_t)(mBase + v) * N + nIdx] = acc[i][j][v];
        }
    }
}

// ---------------------------------------------------------------------------
// BatchNorm stats: one block per channel; biased variance (matches jnp.var)
// ---------------------------------------------------------------------------
__global__ __launch_bounds__(256)
void bnstats_kernel(const float* __restrict__ y, const float* __restrict__ g,
                    const float* __restrict__ beta, float* __restrict__ scale,
                    float* __restrict__ shift, int O) {
    int o = blockIdx.x;
    __shared__ float ssum[256];
    __shared__ float ssq[256];
    float s = 0.f, q = 0.f;
    const int total = BATCH * HWSZ;
    for (int i = threadIdx.x; i < total; i += 256) {
        int b = i / HWSZ, n = i % HWSZ;
        float v = y[((size_t)b * O + o) * HWSZ + n];
        s += v; q += v * v;
    }
    ssum[threadIdx.x] = s; ssq[threadIdx.x] = q;
    __syncthreads();
    for (int st = 128; st > 0; st >>= 1) {
        if (threadIdx.x < st) {
            ssum[threadIdx.x] += ssum[threadIdx.x + st];
            ssq[threadIdx.x]  += ssq[threadIdx.x + st];
        }
        __syncthreads();
    }
    if (threadIdx.x == 0) {
        float mean = ssum[0] / (float)total;
        float var  = ssq[0] / (float)total - mean * mean;
        float sc   = g[o] * rsqrtf(var + 1e-5f);
        scale[o] = sc;
        shift[o] = beta[o] - mean * sc;
    }
}

// ---------------------------------------------------------------------------
// Apply BN (scale/shift) + ReLU in place
// ---------------------------------------------------------------------------
__global__ void bnapply_kernel(float* __restrict__ y, const float* __restrict__ scale,
                               const float* __restrict__ shift, int O, size_t total) {
    size_t idx = (size_t)blockIdx.x * blockDim.x + threadIdx.x;
    if (idx >= total) return;
    int o = (int)((idx / HWSZ) % O);
    float v = fmaf(y[idx], scale[o], shift[o]);
    y[idx] = v > 0.f ? v : 0.f;
}

// ---------------------------------------------------------------------------
// 1x1 attention conv + sigmoid:  att[b][hw]
// ---------------------------------------------------------------------------
__global__ __launch_bounds__(256)
void att_kernel(const float* __restrict__ pA, const float* __restrict__ wA,
                const float* __restrict__ bA, float* __restrict__ att) {
    int idx = blockIdx.x * blockDim.x + threadIdx.x;   // over B*HWSZ
    int b  = idx / HWSZ;
    int hw = idx % HWSZ;
    float acc = bA[0];
    const float* xb = pA + (size_t)b * HCH * HWSZ + hw;
    for (int c = 0; c < HCH; ++c)
        acc = fmaf(xb[(size_t)c * HWSZ], wA[c], acc);
    att[idx] = sigmoidf_(acc);
}

// ---------------------------------------------------------------------------
// out = (2 - att) * y2
// ---------------------------------------------------------------------------
__global__ void final_kernel(const float* __restrict__ y2, const float* __restrict__ att,
                             float* __restrict__ out, size_t total) {
    size_t idx = (size_t)blockIdx.x * blockDim.x + threadIdx.x;
    if (idx >= total) return;
    size_t b  = idx / ((size_t)HCH * HWSZ);
    size_t hw = idx % HWSZ;
    out[idx] = (2.0f - att[b * HWSZ + hw]) * y2[idx];
}

// ---------------------------------------------------------------------------
// Orchestration
// ---------------------------------------------------------------------------
extern "C" void kernel_launch(void* const* d_in, const int* in_sizes, int n_in,
                              void* d_out, int out_size, void* d_ws, size_t ws_size,
                              hipStream_t stream) {
    const float* pA     = (const float*)d_in[0];
    const float* pB     = (const float*)d_in[1];
    const float* off1_w = (const float*)d_in[2];
    const float* off1_b = (const float*)d_in[3];
    const float* w1     = (const float*)d_in[4];
    const float* bn1_g  = (const float*)d_in[5];
    const float* bn1_b  = (const float*)d_in[6];
    const float* off2_w = (const float*)d_in[7];
    const float* off2_b = (const float*)d_in[8];
    const float* w2     = (const float*)d_in[9];
    const float* bn2_g  = (const float*)d_in[10];
    const float* bn2_b  = (const float*)d_in[11];
    const float* attA_w = (const float*)d_in[12];
    const float* attA_b = (const float*)d_in[13];
    float* out = (float*)d_out;

    // workspace carve-up (256B aligned pieces)
    char* p = (char*)d_ws;
    auto take = [&](size_t bytes) {
        char* r = p;
        p += (bytes + 255) & ~(size_t)255;
        return r;
    };
    float*          xcat  = (float*)         take((size_t)BATCH * C2  * HWSZ * 4);
    float*          y1    = (float*)         take((size_t)BATCH * C2  * HWSZ * 4);
    float*          y2    = (float*)         take((size_t)BATCH * HCH * HWSZ * 4);
    float*          om    = (float*)         take((size_t)BATCH * 27  * HWSZ * 4);
    unsigned short* Sbuf  = (unsigned short*)take((size_t)HWSZ * KRED * 2);
    unsigned short* w1bf  = (unsigned short*)take((size_t)C2  * KRED * 2);
    unsigned short* w2bf  = (unsigned short*)take((size_t)HCH * KRED * 2);
    float*          scale = (float*)         take(256 * 4);
    float*          shift = (float*)         take(256 * 4);
    float*          attb  = (float*)         take((size_t)BATCH * HWSZ * 4);

    const size_t totHalf = (size_t)BATCH * HCH * HWSZ;   // 4,718,592
    const size_t totFull = (size_t)BATCH * C2  * HWSZ;

    // 0) concat + weight conversion (with reduction-dim permutation)
    concat_kernel<<<dim3((unsigned)(totHalf / 256)), 256, 0, stream>>>(pA, pB, xcat);
    cvt_w_kernel<<<dim3((C2 * KRED + 255) / 256), 256, 0, stream>>>(w1, w1bf, C2, C2);
    cvt_w_kernel<<<dim3((HCH * KRED + 255) / 256), 256, 0, stream>>>(w2, w2bf, HCH, C2);

    // ---- Layer 1: dfconv(x, off1, w1) -> BN -> ReLU ----
    offconv_kernel<<<dim3(HWSZ / 256, 27, BATCH), 256, 0, stream>>>(xcat, off1_w, off1_b, om, C2);
    for (int b = 0; b < BATCH; ++b) {
        sample_kernel<<<dim3(HWSZ / 256, 9), 256, 0, stream>>>(xcat, om, Sbuf, C2, b);
        gemm_wmma_kernel<<<dim3(HWSZ / BN, C2 / BM), 256, 0, stream>>>(
            w1bf, Sbuf, y1 + (size_t)b * C2 * HWSZ, C2, KRED, HWSZ);
    }
    bnstats_kernel<<<dim3(C2), 256, 0, stream>>>(y1, bn1_g, bn1_b, scale, shift, C2);
    bnapply_kernel<<<dim3((unsigned)((totFull + 255) / 256)), 256, 0, stream>>>(
        y1, scale, shift, C2, totFull);

    // ---- Layer 2: dfconv(y1, off2, w2) -> BN -> ReLU ----
    offconv_kernel<<<dim3(HWSZ / 256, 27, BATCH), 256, 0, stream>>>(y1, off2_w, off2_b, om, C2);
    for (int b = 0; b < BATCH; ++b) {
        sample_kernel<<<dim3(HWSZ / 256, 9), 256, 0, stream>>>(y1, om, Sbuf, C2, b);
        gemm_wmma_kernel<<<dim3(HWSZ / BN, HCH / BM), 256, 0, stream>>>(
            w2bf, Sbuf, y2 + (size_t)b * HCH * HWSZ, HCH, KRED, HWSZ);
    }
    bnstats_kernel<<<dim3(HCH), 256, 0, stream>>>(y2, bn2_g, bn2_b, scale, shift, HCH);
    bnapply_kernel<<<dim3((unsigned)((totHalf + 255) / 256)), 256, 0, stream>>>(
        y2, scale, shift, HCH, totHalf);

    // ---- Attention + final gate ----
    att_kernel<<<dim3(BATCH * HWSZ / 256), 256, 0, stream>>>(pA, attA_w, attA_b, attb);
    final_kernel<<<dim3((unsigned)((totHalf + 255) / 256)), 256, 0, stream>>>(
        y2, attb, out, totHalf);
}
